// CrossAttentionBlock_41429254537586
// MI455X (gfx1250) — matrix operand order repaired and checked by reading, not compiled, not run on hardware
//
#include <hip/hip_runtime.h>

// ---------------------------------------------------------------------------
// CDNA5 (gfx1250) cross-attention block: bf16 WMMA GEMMs + flash attention.
// Data movement via Tensor Data Mover (TENSOR_LOAD_TO_LDS, TENSORcnt
// double-buffering).  Matrix ops via v_wmma_f32_16x16x32_bf16.
// ---------------------------------------------------------------------------

typedef __attribute__((ext_vector_type(16))) __bf16 v16bf;
typedef __attribute__((ext_vector_type(8)))  float  v8f;
typedef __attribute__((ext_vector_type(4)))  unsigned int v4u;
typedef __attribute__((ext_vector_type(8)))  int v8i;
typedef __attribute__((ext_vector_type(4)))  int v4i;

#define D_MODEL 1024
#define NHEAD   16
#define HDIM    64
#define SEQLEN  1024
#define BATCH   8
#define ROWS    (BATCH * SEQLEN)   // 8192
#define FFDIM   4096
#define ATT_SCALE 0.125f           // HDIM^-0.5

// ----------------------------- bf16 helpers --------------------------------
__device__ __forceinline__ unsigned short f32_to_bf16(float f) {
  unsigned u = __builtin_bit_cast(unsigned, f);
  u += 0x7FFFu + ((u >> 16) & 1u);            // round-to-nearest-even
  return (unsigned short)(u >> 16);
}
__device__ __forceinline__ float bf16_to_f32(unsigned short h) {
  unsigned u = ((unsigned)h) << 16;
  return __builtin_bit_cast(float, u);
}

// ------------------- Tensor Data Mover (CDNA5) staging ---------------------
// 2-D bf16 tile load: tile_w elems per row (64 here -> 32 DWORDs/row),
// LDS rows padded by 4 DWORDs (8 elems) => LDS row stride 72 elems.
// D# layout per CDNA5 ISA; 6-arg builtin form.
__device__ __forceinline__ void tdm_load_2d(unsigned lds_addr, const void* gptr,
                                            unsigned tile_w, unsigned tile_h,
                                            unsigned row_stride_elems) {
  const unsigned long long ga = (unsigned long long)gptr;
  v4u g0;
  g0[0] = 1u;                                   // count=1, user-mode load
  g0[1] = lds_addr;                             // LDS byte address
  g0[2] = (unsigned)(ga & 0xFFFFFFFFu);         // global_addr[31:0]
  g0[3] = (unsigned)((ga >> 32) & 0x01FFFFFFu)  // global_addr[56:32]
          | (2u << 30);                         // type=2 ("image")
  v8i g1;
  g1[0] = (int)((1u << 16)      // data_size = 2 bytes
                | (1u << 20)    // pad_enable
                | (4u << 22)    // pad_interval: every 32 DWORDs
                | (3u << 25));  // pad_amount: 4 DWORDs
  g1[1] = (int)((tile_w & 0xFFFFu) << 16);                      // tensor_dim0 lo
  g1[2] = (int)(((tile_w >> 16) & 0xFFFFu) |
                ((tile_h & 0xFFFFu) << 16));                    // dim0 hi | dim1 lo
  g1[3] = (int)(((tile_h >> 16) & 0xFFFFu) |
                ((tile_w & 0xFFFFu) << 16));                    // dim1 hi | tile_dim0
  g1[4] = (int)(tile_h & 0xFFFFu);                              // tile_dim1 (dim2=0)
  g1[5] = (int)row_stride_elems;                                // dim0_stride lo32
  g1[6] = 0;
  g1[7] = 0;
  const v4i z4 = {0, 0, 0, 0};                  // 2-D: groups 2/3 unused
  const v8i z8 = {0, 0, 0, 0, 0, 0, 0, 0};
  __builtin_amdgcn_tensor_load_to_lds(g0, g1, z4, z4, z8, 0);
}
__device__ __forceinline__ unsigned lds_off(const void* p) {
  return (unsigned)(size_t)p;   // LDS byte offset = addr[31:0] (aperture rule)
}

// ----------------------- WMMA fragment loaders (LDS) -----------------------
// A fragment 16x32 bf16 (this file's contiguous k order, same as B): lane r=lane&15 holds row r;
// element e -> vgpr j=e/2, pair p=e&1: K = 2j+p + 16*(lane>>4)
__device__ __forceinline__ v16bf load_frag_a(const __bf16* smem, int stride, int lane) {
  const int r = lane & 15, hh = lane >> 4;
  v16bf f;
#pragma unroll
  for (int e = 0; e < 16; ++e) {
    const int j = e >> 1, p = e & 1;
    const int kk = 2 * j + p + 16 * hh;
    f[e] = smem[r * stride + kk];
  }
  return f;
}
// B fragment 32x16 bf16 from LDS stored [K][N]: K = 2j+p+16*(lane>>4), N=lane&15
__device__ __forceinline__ v16bf load_frag_b(const __bf16* smem, int stride, int lane) {
  const int n = lane & 15, hh = lane >> 4;
  v16bf f;
#pragma unroll
  for (int e = 0; e < 16; ++e) {
    const int j = e >> 1, p = e & 1;
    const int kk = 2 * j + p + 16 * hh;
    f[e] = smem[kk * stride + n];
  }
  return f;
}
// B fragment from LDS stored transposed [N][K] (for Q@K^T, K row-major)
__device__ __forceinline__ v16bf load_frag_bt(const __bf16* smem, int stride, int lane) {
  const int n = lane & 15, hh = lane >> 4;
  v16bf f;
#pragma unroll
  for (int e = 0; e < 16; ++e) {
    const int j = e >> 1, p = e & 1;
    const int kk = 2 * j + p + 16 * hh;
    f[e] = smem[n * stride + kk];
  }
  return f;
}
__device__ __forceinline__ v8f wmma_bf16(v16bf a, v16bf b, v8f c) {
  return __builtin_amdgcn_wmma_f32_16x16x32_bf16(false, a, false, b, (short)0, c,
                                                 false, false);
}

// --------------------- f32 -> bf16 convert kernel --------------------------
__global__ __launch_bounds__(256) void cvt_bf16_kernel(
    const float* __restrict__ in, unsigned short* __restrict__ out, int n) {
  const int i = (blockIdx.x * 256 + threadIdx.x) * 8;
  if (i + 8 > n) return;
  const float4 a = ((const float4*)(in + i))[0];
  const float4 b = ((const float4*)(in + i))[1];
  uint4 o;
  o.x = (unsigned)f32_to_bf16(a.x) | ((unsigned)f32_to_bf16(a.y) << 16);
  o.y = (unsigned)f32_to_bf16(a.z) | ((unsigned)f32_to_bf16(a.w) << 16);
  o.z = (unsigned)f32_to_bf16(b.x) | ((unsigned)f32_to_bf16(b.y) << 16);
  o.w = (unsigned)f32_to_bf16(b.z) | ((unsigned)f32_to_bf16(b.w) << 16);
  *(uint4*)(out + i) = o;
}

// ------------------------------ GEMM kernel --------------------------------
// C[M,N](bf16) = epilogue(A[M,K](bf16) @ W[K,N](bf16) + bias(f32)),
// 128x64 block tile, 8 waves of 32x32, K-step 64, TDM double-buffered staging.
// RES: 0 none, 1 +f32 residual, 2 +bf16 residual.
#define GT_M 128
#define GT_N 64
#define GT_K 64

template <bool RELU, int RES>
__global__ __launch_bounds__(256) void gemm_kernel(
    const unsigned short* __restrict__ A, const unsigned short* __restrict__ W,
    const float* __restrict__ bias, const float* __restrict__ resF,
    const unsigned short* __restrict__ resB, unsigned short* __restrict__ Cout,
    int N, int K) {
  __shared__ __bf16 sA[2][GT_M][GT_K + 8];
  __shared__ __bf16 sB[2][GT_K][GT_N + 8];

  const int tid  = threadIdx.x;
  const int lane = tid & 31;
  const int wave = tid >> 5;
  const int wm   = wave & 3;   // 4 waves along M
  const int wn   = wave >> 2;  // 2 waves along N
  const int bm   = blockIdx.x * GT_M;
  const int bn   = blockIdx.y * GT_N;

  v8f acc[2][2] = {};

  if (wave == 0) {  // prologue: TDM load of first tiles into buffer 0
    tdm_load_2d(lds_off(&sA[0][0][0]), A + (size_t)bm * K, GT_K, GT_M, K);
    tdm_load_2d(lds_off(&sB[0][0][0]), W + bn, GT_N, GT_K, N);
  }

  const int nIter = K / GT_K;
  for (int it = 0; it < nIter; ++it) {
    const int cur = it & 1;
    const int k0  = it * GT_K;
    if (wave == 0) {
      if (it + 1 < nIter) {  // issue next tiles into other buffer, keep 2 in flight
        tdm_load_2d(lds_off(&sA[cur ^ 1][0][0]),
                    A + (size_t)bm * K + (k0 + GT_K), GT_K, GT_M, K);
        tdm_load_2d(lds_off(&sB[cur ^ 1][0][0]),
                    W + (size_t)(k0 + GT_K) * N + bn, GT_N, GT_K, N);
        __builtin_amdgcn_s_wait_tensorcnt(2);
      } else {
        __builtin_amdgcn_s_wait_tensorcnt(0);
      }
    }
    __syncthreads();

    v16bf af[2][2], bfr[2][2];
#pragma unroll
    for (int i = 0; i < 2; ++i)
#pragma unroll
      for (int kc = 0; kc < 2; ++kc)
        af[i][kc] =
            load_frag_a(&sA[cur][wm * 32 + i * 16][kc * 32], GT_K + 8, lane);
#pragma unroll
    for (int kc = 0; kc < 2; ++kc)
#pragma unroll
      for (int j = 0; j < 2; ++j)
        bfr[kc][j] =
            load_frag_b(&sB[cur][kc * 32][wn * 32 + j * 16], GT_N + 8, lane);
#pragma unroll
    for (int i = 0; i < 2; ++i)
#pragma unroll
      for (int j = 0; j < 2; ++j)
#pragma unroll
        for (int kc = 0; kc < 2; ++kc)
          acc[i][j] = wmma_bf16(af[i][kc], bfr[kc][j], acc[i][j]);
    __syncthreads();
  }

  // epilogue: C lane layout row = 8*(lane>>4)+v, col = lane&15
  const int hh = lane >> 4, nn = lane & 15;
#pragma unroll
  for (int i = 0; i < 2; ++i) {
#pragma unroll
    for (int j = 0; j < 2; ++j) {
      const int colg = bn + wn * 32 + j * 16 + nn;
      const float bv = bias[colg];
#pragma unroll
      for (int v = 0; v < 8; ++v) {
        const int rowg = bm + wm * 32 + i * 16 + 8 * hh + v;
        float val = acc[i][j][v] + bv;
        if (RELU) val = fmaxf(val, 0.0f);
        if (RES == 1) val += resF[(size_t)rowg * N + colg];
        if (RES == 2) val += bf16_to_f32(resB[(size_t)rowg * N + colg]);
        Cout[(size_t)rowg * N + colg] = f32_to_bf16(val);
      }
    }
  }
}

// --------------------------- all-pad prep kernel ---------------------------
__global__ __launch_bounds__(256) void allpad_kernel(
    const unsigned char* __restrict__ mask, int* __restrict__ allpad) {
  __shared__ int red[256];
  const int b = blockIdx.x, tid = threadIdx.x;
  int ok = 1;
  for (int i = tid; i < SEQLEN; i += 256) ok &= (mask[b * SEQLEN + i] ? 1 : 0);
  red[tid] = ok;
  __syncthreads();
  for (int s = 128; s > 0; s >>= 1) {
    if (tid < s) red[tid] &= red[tid + s];
    __syncthreads();
  }
  if (tid == 0) allpad[b] = red[0];
}

// --------------------------- flash attention -------------------------------
// grid (B*H, Lq/64), block 128 (4 waves).  Each wave: 16 q-rows; keys in
// blocks of 64 staged by TDM (double-buffered): S = Q@K^T (8 WMMA),
// online softmax, O += P@V (8 WMMA).
#define KBLK 64

__global__ __launch_bounds__(128) void attn_kernel(
    const unsigned short* __restrict__ qp, const unsigned short* __restrict__ kp,
    const unsigned short* __restrict__ vp, const unsigned char* __restrict__ mask,
    const int* __restrict__ allpad, unsigned short* __restrict__ ctx) {
  const int bh   = blockIdx.x;
  const int b    = bh >> 4;  // /NHEAD
  const int h    = bh & 15;
  const int tid  = threadIdx.x;
  const int lane = tid & 31;
  const int wave = tid >> 5;
  const int q0   = blockIdx.y * 64;

  __shared__ __bf16 sQ[64][HDIM + 8];
  __shared__ __bf16 sK[2][KBLK][HDIM + 8];
  __shared__ __bf16 sV[2][KBLK][HDIM + 8];
  __shared__ __bf16 sP[4][16][KBLK + 8];

  const unsigned short* qbase = qp + ((size_t)(b * SEQLEN + q0)) * D_MODEL + h * HDIM;
  const unsigned short* kbase = kp + ((size_t)(b * SEQLEN)) * D_MODEL + h * HDIM;
  const unsigned short* vbase = vp + ((size_t)(b * SEQLEN)) * D_MODEL + h * HDIM;

  if (wave == 0) {  // TDM: Q tile first
    tdm_load_2d(lds_off(&sQ[0][0]), qbase, HDIM, 64, D_MODEL);
    __builtin_amdgcn_s_wait_tensorcnt(0);
  }
  __syncthreads();
  v16bf qa[2];
  qa[0] = load_frag_a(&sQ[wave * 16][0], HDIM + 8, lane);
  qa[1] = load_frag_a(&sQ[wave * 16][32], HDIM + 8, lane);
  if (wave == 0) {  // first K/V tiles into buffer 0
    tdm_load_2d(lds_off(&sK[0][0][0]), kbase, HDIM, KBLK, D_MODEL);
    tdm_load_2d(lds_off(&sV[0][0][0]), vbase, HDIM, KBLK, D_MODEL);
  }

  v8f oc[4] = {};
  float mrow[8], lrow[8];
#pragma unroll
  for (int v = 0; v < 8; ++v) { mrow[v] = -3.0e38f; lrow[v] = 0.0f; }

  const int apb = allpad[b];
  const int hh = lane >> 4, nn = lane & 15;

  const int nIter = SEQLEN / KBLK;
  for (int it = 0; it < nIter; ++it) {
    const int cur = it & 1;
    const int kb  = it * KBLK;
    if (wave == 0) {
      if (it + 1 < nIter) {
        const size_t nb = (size_t)(kb + KBLK) * D_MODEL;
        tdm_load_2d(lds_off(&sK[cur ^ 1][0][0]), kbase + nb, HDIM, KBLK, D_MODEL);
        tdm_load_2d(lds_off(&sV[cur ^ 1][0][0]), vbase + nb, HDIM, KBLK, D_MODEL);
        __builtin_amdgcn_s_wait_tensorcnt(2);
      } else {
        __builtin_amdgcn_s_wait_tensorcnt(0);
      }
    }
    __syncthreads();

    // S = Q @ K^T  (four 16-key groups, contraction over HDIM=64)
    v8f sc[4] = {};
#pragma unroll
    for (int g = 0; g < 4; ++g)
#pragma unroll
      for (int kc = 0; kc < 2; ++kc)
        sc[g] = wmma_bf16(
            qa[kc], load_frag_bt(&sK[cur][g * 16][kc * 32], HDIM + 8, lane), sc[g]);

    // scale + mask + online softmax (rows live in 16-lane halves)
    float pv[4][8], corr[8];
#pragma unroll
    for (int v = 0; v < 8; ++v) {
#pragma unroll
      for (int g = 0; g < 4; ++g) {
        float s = sc[g][v] * ATT_SCALE;
        const int kidx = kb + g * 16 + nn;
        const bool pad = mask[b * SEQLEN + kidx] != 0;
        if (pad && !(apb && kidx == SEQLEN - 1)) s = -10000.0f;
        sc[g][v] = s;
      }
      float mx = fmaxf(fmaxf(sc[0][v], sc[1][v]), fmaxf(sc[2][v], sc[3][v]));
#pragma unroll
      for (int off = 1; off < 16; off <<= 1) mx = fmaxf(mx, __shfl_xor(mx, off, 32));
      const float m_new = fmaxf(mrow[v], mx);
      corr[v] = __expf(mrow[v] - m_new);
      float rs = 0.0f;
#pragma unroll
      for (int g = 0; g < 4; ++g) {
        const float p = __expf(sc[g][v] - m_new);
        pv[g][v] = p;
        rs += p;
      }
#pragma unroll
      for (int off = 1; off < 16; off <<= 1) rs += __shfl_xor(rs, off, 32);
      lrow[v] = lrow[v] * corr[v] + rs;
      mrow[v] = m_new;
    }
#pragma unroll
    for (int j = 0; j < 4; ++j)
#pragma unroll
      for (int v = 0; v < 8; ++v) oc[j][v] *= corr[v];

    // reshape P (C layout) -> A layout via per-wave LDS scratch
#pragma unroll
    for (int g = 0; g < 4; ++g)
#pragma unroll
      for (int v = 0; v < 8; ++v)
        sP[wave][8 * hh + v][g * 16 + nn] =
            __builtin_bit_cast(__bf16, f32_to_bf16(pv[g][v]));
    asm volatile("" ::: "memory");  // keep LDS store->load order (same wave)
    v16bf pa[2];
    pa[0] = load_frag_a(&sP[wave][0][0], KBLK + 8, lane);
    pa[1] = load_frag_a(&sP[wave][0][32], KBLK + 8, lane);

    // O += P @ V  (contraction over 64 keys)
#pragma unroll
    for (int j = 0; j < 4; ++j)
#pragma unroll
      for (int kc = 0; kc < 2; ++kc)
        oc[j] = wmma_bf16(
            pa[kc], load_frag_b(&sV[cur][kc * 32][j * 16], HDIM + 8, lane), oc[j]);
    __syncthreads();
  }

  // normalize and write context
#pragma unroll
  for (int j = 0; j < 4; ++j) {
#pragma unroll
    for (int v = 0; v < 8; ++v) {
      const int qrow = q0 + wave * 16 + 8 * hh + v;
      const int d    = j * 16 + nn;
      ctx[((size_t)(b * SEQLEN + qrow)) * D_MODEL + h * HDIM + d] =
          f32_to_bf16(oc[j][v] / lrow[v]);
    }
  }
}

// ------------------------------ LayerNorm ----------------------------------
template <bool OUT_F32>
__global__ __launch_bounds__(256) void ln_kernel(
    const unsigned short* __restrict__ in, const float* __restrict__ gamma,
    const float* __restrict__ beta, unsigned short* __restrict__ outB,
    float* __restrict__ outF) {
  __shared__ float red[256];
  const int row = blockIdx.x, tid = threadIdx.x;
  float vals[4];
  {
    const uint2 v = *(const uint2*)(in + (size_t)row * D_MODEL + tid * 4);
    vals[0] = bf16_to_f32((unsigned short)(v.x & 0xFFFFu));
    vals[1] = bf16_to_f32((unsigned short)(v.x >> 16));
    vals[2] = bf16_to_f32((unsigned short)(v.y & 0xFFFFu));
    vals[3] = bf16_to_f32((unsigned short)(v.y >> 16));
  }
  red[tid] = vals[0] + vals[1] + vals[2] + vals[3];
  __syncthreads();
  for (int st = 128; st > 0; st >>= 1) {
    if (tid < st) red[tid] += red[tid + st];
    __syncthreads();
  }
  const float mean = red[0] * (1.0f / D_MODEL);
  __syncthreads();
  float sq = 0.0f;
#pragma unroll
  for (int i = 0; i < 4; ++i) {
    const float d = vals[i] - mean;
    sq += d * d;
  }
  red[tid] = sq;
  __syncthreads();
  for (int st = 128; st > 0; st >>= 1) {
    if (tid < st) red[tid] += red[tid + st];
    __syncthreads();
  }
  const float rstd = rsqrtf(red[0] * (1.0f / D_MODEL) + 1e-5f);
#pragma unroll
  for (int i = 0; i < 4; ++i) {
    const int col = tid * 4 + i;
    const float y = (vals[i] - mean) * rstd * gamma[col] + beta[col];
    if (OUT_F32)
      outF[(size_t)row * D_MODEL + col] = y;
    else
      outB[(size_t)row * D_MODEL + col] = f32_to_bf16(y);
  }
}

// ------------------------------ launcher -----------------------------------
extern "C" void kernel_launch(void* const* d_in, const int* in_sizes, int n_in,
                              void* d_out, int out_size, void* d_ws, size_t ws_size,
                              hipStream_t stream) {
  const float* q   = (const float*)d_in[0];
  const float* kv  = (const float*)d_in[1];
  const unsigned char* kvm = (const unsigned char*)d_in[2];
  const float* Wq = (const float*)d_in[3];
  const float* bq = (const float*)d_in[4];
  const float* Wk = (const float*)d_in[5];
  const float* bk = (const float*)d_in[6];
  const float* Wv = (const float*)d_in[7];
  const float* bv = (const float*)d_in[8];
  const float* Wo = (const float*)d_in[9];
  const float* bo = (const float*)d_in[10];
  const float* g1 = (const float*)d_in[11];
  const float* be1 = (const float*)d_in[12];
  const float* W1 = (const float*)d_in[13];
  const float* b1 = (const float*)d_in[14];
  const float* W2 = (const float*)d_in[15];
  const float* b2 = (const float*)d_in[16];
  const float* g2 = (const float*)d_in[17];
  const float* be2 = (const float*)d_in[18];
  float* out = (float*)d_out;

  // workspace layout (bf16 buffers as ushort)
  unsigned short* qbf  = (unsigned short*)d_ws;
  unsigned short* kvbf = qbf + (size_t)ROWS * D_MODEL;
  unsigned short* wqb  = kvbf + (size_t)ROWS * D_MODEL;
  unsigned short* wkb  = wqb + (size_t)D_MODEL * D_MODEL;
  unsigned short* wvb  = wkb + (size_t)D_MODEL * D_MODEL;
  unsigned short* wob  = wvb + (size_t)D_MODEL * D_MODEL;
  unsigned short* w1b  = wob + (size_t)D_MODEL * D_MODEL;
  unsigned short* w2b  = w1b + (size_t)D_MODEL * FFDIM;
  unsigned short* qp   = w2b + (size_t)FFDIM * D_MODEL;
  unsigned short* kp   = qp + (size_t)ROWS * D_MODEL;
  unsigned short* vp   = kp + (size_t)ROWS * D_MODEL;
  unsigned short* ctx  = vp + (size_t)ROWS * D_MODEL;
  unsigned short* s1   = ctx + (size_t)ROWS * D_MODEL;
  unsigned short* xb   = s1 + (size_t)ROWS * D_MODEL;
  unsigned short* hb   = xb + (size_t)ROWS * D_MODEL;
  unsigned short* s2   = hb + (size_t)ROWS * FFDIM;
  int* allpad          = (int*)(s2 + (size_t)ROWS * D_MODEL);

  const dim3 blk256(256), blk128(128);
  const dim3 gProj(ROWS / GT_M, D_MODEL / GT_N);   // (64,16)
  const dim3 gFF1(ROWS / GT_M, FFDIM / GT_N);      // (64,64)
  const dim3 gAttn(BATCH * NHEAD, SEQLEN / 64);    // (128,16)

  // one-shot f32 -> bf16 conversions (activations + weights)
  const int nDD = D_MODEL * D_MODEL, nDF = D_MODEL * FFDIM, nRD = ROWS * D_MODEL;
  cvt_bf16_kernel<<<dim3(nRD / 2048), blk256, 0, stream>>>(q, qbf, nRD);
  cvt_bf16_kernel<<<dim3(nRD / 2048), blk256, 0, stream>>>(kv, kvbf, nRD);
  cvt_bf16_kernel<<<dim3(nDD / 2048), blk256, 0, stream>>>(Wq, wqb, nDD);
  cvt_bf16_kernel<<<dim3(nDD / 2048), blk256, 0, stream>>>(Wk, wkb, nDD);
  cvt_bf16_kernel<<<dim3(nDD / 2048), blk256, 0, stream>>>(Wv, wvb, nDD);
  cvt_bf16_kernel<<<dim3(nDD / 2048), blk256, 0, stream>>>(Wo, wob, nDD);
  cvt_bf16_kernel<<<dim3(nDF / 2048), blk256, 0, stream>>>(W1, w1b, nDF);
  cvt_bf16_kernel<<<dim3(nDF / 2048), blk256, 0, stream>>>(W2, w2b, nDF);

  allpad_kernel<<<dim3(BATCH), blk256, 0, stream>>>(kvm, allpad);

  // Q/K/V projections
  gemm_kernel<false, 0><<<gProj, blk256, 0, stream>>>(qbf, wqb, bq, nullptr, nullptr, qp, D_MODEL, D_MODEL);
  gemm_kernel<false, 0><<<gProj, blk256, 0, stream>>>(kvbf, wkb, bk, nullptr, nullptr, kp, D_MODEL, D_MODEL);
  gemm_kernel<false, 0><<<gProj, blk256, 0, stream>>>(kvbf, wvb, bv, nullptr, nullptr, vp, D_MODEL, D_MODEL);

  // attention
  attn_kernel<<<gAttn, blk128, 0, stream>>>(qp, kp, vp, kvm, allpad, ctx);

  // output projection + residual(q, f32) -> s1; LN1 -> x (bf16)
  gemm_kernel<false, 1><<<gProj, blk256, 0, stream>>>(ctx, wob, bo, q, nullptr, s1, D_MODEL, D_MODEL);
  ln_kernel<false><<<dim3(ROWS), blk256, 0, stream>>>(s1, g1, be1, xb, nullptr);

  // FFN: relu(x@W1+b1) -> h ; h@W2+b2 + x -> s2 ; LN2 -> out (f32)
  gemm_kernel<true, 0><<<gFF1, blk256, 0, stream>>>(xb, w1b, b1, nullptr, nullptr, hb, FFDIM, D_MODEL);
  gemm_kernel<false, 2><<<gProj, blk256, 0, stream>>>(hb, w2b, b2, nullptr, xb, s2, D_MODEL, FFDIM);
  ln_kernel<true><<<dim3(ROWS), blk256, 0, stream>>>(s2, g2, be2, nullptr, out);
}
